// CausalSelfAttention_87342454932023
// MI455X (gfx1250) — matrix operand order repaired and checked by reading, not compiled, not run on hardware
//
#include <hip/hip_runtime.h>

// ---------------------------------------------------------------------------
// Problem constants (B=2, T=2048, D=1024, H=16, DH=64)
// ---------------------------------------------------------------------------
#define B_   2
#define T_   2048
#define D_   1024
#define H_   16
#define DH_  64
#define TD3  3072           // 3*D
#define MROWS 4096          // B*T

typedef __attribute__((ext_vector_type(16))) __bf16 v16bf;
typedef __attribute__((ext_vector_type(8)))  __bf16 v8bf;
typedef __attribute__((ext_vector_type(8)))  float  v8f;

union BFrag { v16bf v; v8bf h[2]; };

__device__ __forceinline__ v8bf ld8(const __bf16* p) { return *(const v8bf*)p; }

#define WMMA_BF16(A, Bv, C) \
    __builtin_amdgcn_wmma_f32_16x16x32_bf16(false, (A), false, (Bv), (short)0, (C), false, false)

// ---------------------------------------------------------------------------
// f32 -> bf16 convert
// ---------------------------------------------------------------------------
__global__ __launch_bounds__(256) void cvt_f32_bf16(const float* __restrict__ in,
                                                    __bf16* __restrict__ out, int n) {
    int i = blockIdx.x * 256 + threadIdx.x;
    if (i < n) out[i] = (__bf16)in[i];
}

// w [K,N] f32  ->  wT [N,K] bf16
__global__ __launch_bounds__(256) void transpose_cvt(const float* __restrict__ w,
                                                     __bf16* __restrict__ wt,
                                                     int K, int N) {
    int idx = blockIdx.x * 256 + threadIdx.x;
    if (idx >= K * N) return;
    int n = idx % N;
    int k = idx / N;
    wt[(size_t)n * K + k] = (__bf16)w[(size_t)k * N + n];
}

// ---------------------------------------------------------------------------
// Generic bf16 GEMM: C[M,N] = A[M,K] * B[K,N], B given pre-transposed as Bt[N,K].
// Block = 256 threads = 8 waves; block tile 64(M) x 128(N); wave tile 16 x 64.
// Grid = (N/128, M/64). K-loop double-buffered so WMMAs never wait on the
// loads issued in the same stage. Stores bf16 (Cb) or f32 (Cf).
// ---------------------------------------------------------------------------
struct KStep { BFrag a; BFrag b[4]; };

__device__ __forceinline__ void load_kstep(KStep& f, const __bf16* arow,
                                           const __bf16* const brow[4],
                                           int kk, int half) {
    f.a.h[0] = ld8(arow + kk + half * 8);
    f.a.h[1] = ld8(arow + kk + 16 + half * 8);
#pragma unroll
    for (int j = 0; j < 4; ++j) {
        f.b[j].h[0] = ld8(brow[j] + kk + half * 8);
        f.b[j].h[1] = ld8(brow[j] + kk + 16 + half * 8);
    }
}

__global__ __launch_bounds__(256)
void gemm_bf16_wmma(const __bf16* __restrict__ A, const __bf16* __restrict__ Bt,
                    __bf16* __restrict__ Cb, float* __restrict__ Cf,
                    int K, int N) {
    const int lane = threadIdx.x & 31;
    const int w    = threadIdx.x >> 5;
    const int half = lane >> 4;
    const int ln   = lane & 15;
    const int wm   = w & 3;
    const int wn   = w >> 2;
    const int m0   = blockIdx.y * 64 + wm * 16;
    const int n0   = blockIdx.x * 128 + wn * 64;

    const __bf16* arow = A + (size_t)(m0 + ln) * K;
    const __bf16* brow[4];
#pragma unroll
    for (int j = 0; j < 4; ++j)
        brow[j] = Bt + (size_t)(n0 + j * 16 + ln) * K;

    v8f acc[4] = {v8f{}, v8f{}, v8f{}, v8f{}};

    KStep f0, f1;
    load_kstep(f0, arow, brow, 0, half);
    for (int kk = 0; kk < K; kk += 64) {          // K multiple of 64
        load_kstep(f1, arow, brow, kk + 32, half);
        __builtin_prefetch(arow + kk + 256, 0, 1);
#pragma unroll
        for (int j = 0; j < 4; ++j)
            acc[j] = WMMA_BF16(f0.a.v, f0.b[j].v, acc[j]);
        if (kk + 64 < K)
            load_kstep(f0, arow, brow, kk + 64, half);
#pragma unroll
        for (int j = 0; j < 4; ++j)
            acc[j] = WMMA_BF16(f1.a.v, f1.b[j].v, acc[j]);
    }

    // C layout: VGPR i -> row m0 + i + half*8 ; col = n0 + j*16 + ln
    if (Cf) {
#pragma unroll
        for (int j = 0; j < 4; ++j)
#pragma unroll
            for (int i = 0; i < 8; ++i)
                Cf[(size_t)(m0 + i + half * 8) * N + n0 + j * 16 + ln] = acc[j][i];
    } else {
#pragma unroll
        for (int j = 0; j < 4; ++j)
#pragma unroll
            for (int i = 0; i < 8; ++i)
                Cb[(size_t)(m0 + i + half * 8) * N + n0 + j * 16 + ln] =
                    (__bf16)acc[j][i];
    }
}

// ---------------------------------------------------------------------------
// RoPE applied in place to q and k inside qkv bf16 buffer [B*T, 3D].
// One thread per (b,t,h,i) with i in [0,32); rotates pair (i, i+32).
// ---------------------------------------------------------------------------
__global__ __launch_bounds__(256) void rope_kernel(__bf16* __restrict__ qkv) {
    int idx = blockIdx.x * 256 + threadIdx.x;
    int i = idx & 31;
    int h = (idx >> 5) & (H_ - 1);
    int t = (idx >> 9) & (T_ - 1);
    int b = idx >> 20;
    // inv_freq = 10000^(-2i/64)
    float inv_freq = __expf(-(float)(2 * i) * (1.0f / 64.0f) * 9.210340371976184f);
    float ang = (float)t * inv_freq;
    float s, c;
    __sincosf(ang, &s, &c);
    __bf16* row = qkv + (size_t)(b * T_ + t) * TD3 + h * DH_;
    float q0 = (float)row[i], q1 = (float)row[i + 32];
    row[i]      = (__bf16)(q0 * c - q1 * s);
    row[i + 32] = (__bf16)(q1 * c + q0 * s);
    float k0 = (float)row[D_ + i], k1 = (float)row[D_ + i + 32];
    row[D_ + i]      = (__bf16)(k0 * c - k1 * s);
    row[D_ + i + 32] = (__bf16)(k1 * c + k0 * s);
}

// V [B,T,H,DH] (inside qkv) -> Vt [B,H,DH,T] so PV B-fragments are contiguous in s.
__global__ __launch_bounds__(256) void vtrans_kernel(const __bf16* __restrict__ qkv,
                                                     __bf16* __restrict__ vt) {
    int idx = blockIdx.x * 256 + threadIdx.x;
    int t  = idx & (T_ - 1);
    int dh = (idx >> 11) & (DH_ - 1);
    int bh = idx >> 17;
    int b = bh >> 4, h = bh & (H_ - 1);
    vt[((size_t)bh * DH_ + dh) * T_ + t] =
        qkv[(size_t)(b * T_ + t) * TD3 + 2 * D_ + h * DH_ + dh];
}

// ---------------------------------------------------------------------------
// Flash-style causal attention.
// Grid (T/64, B*H), block 128 = 4 waves; wave owns 16 query rows.
// Key tiles of 64: 8 WMMA for S, 8 WMMA for PV per tile; online softmax with
// half-group shuffle reductions; P converted C-layout -> A-layout via a
// per-wave private LDS tile (DS ops are in-order within a wave).
// ---------------------------------------------------------------------------
__global__ __launch_bounds__(128)
void attn_kernel(const __bf16* __restrict__ qkv, const __bf16* __restrict__ vt,
                 __bf16* __restrict__ attn) {
    __shared__ __align__(16) __bf16 plds[4][16][64];

    const int lane = threadIdx.x & 31;
    const int w    = threadIdx.x >> 5;
    const int half = lane >> 4;
    const int ln   = lane & 15;
    const int bh   = blockIdx.y;
    const int b    = bh >> 4;
    const int h    = bh & (H_ - 1);
    const int t0   = blockIdx.x * 64 + w * 16;

    // Q A-fragments for dh 0..31 and 32..63 (loaded once)
    const __bf16* qrow = qkv + (size_t)(b * T_ + t0 + ln) * TD3 + h * DH_;
    BFrag aq0, aq1;
    aq0.h[0] = ld8(qrow + half * 8);
    aq0.h[1] = ld8(qrow + 16 + half * 8);
    aq1.h[0] = ld8(qrow + 32 + half * 8);
    aq1.h[1] = ld8(qrow + 48 + half * 8);

    float m_r[8], l_r[8];
#pragma unroll
    for (int i = 0; i < 8; ++i) { m_r[i] = -3.0e38f; l_r[i] = 0.0f; }
    v8f o[4] = {v8f{}, v8f{}, v8f{}, v8f{}};

    const int tmax = t0 + 15;
    for (int s0 = 0; s0 <= tmax; s0 += 64) {
        // ---- load all K B-fragments for this 64-key tile as one batch
        BFrag bk[4][2];
#pragma unroll
        for (int j = 0; j < 4; ++j) {
            const __bf16* krow =
                qkv + (size_t)(b * T_ + s0 + j * 16 + ln) * TD3 + D_ + h * DH_;
            bk[j][0].h[0] = ld8(krow + half * 8);
            bk[j][0].h[1] = ld8(krow + 16 + half * 8);
            bk[j][1].h[0] = ld8(krow + 32 + half * 8);
            bk[j][1].h[1] = ld8(krow + 48 + half * 8);
        }

        // ---- S = Q * K^T : 16x64 score tile
        v8f sc[4];
#pragma unroll
        for (int j = 0; j < 4; ++j) {
            v8f z = {};
            z = WMMA_BF16(aq0.v, bk[j][0].v, z);
            sc[j] = WMMA_BF16(aq1.v, bk[j][1].v, z);
        }

        // ---- prefetch all V B-fragments now; latency hides under softmax VALU
        BFrag bv0[4], bv1[4];
#pragma unroll
        for (int jt = 0; jt < 4; ++jt) {
            const __bf16* vrow = vt + ((size_t)bh * DH_ + jt * 16 + ln) * T_;
            bv0[jt].h[0] = ld8(vrow + s0 + half * 8);
            bv0[jt].h[1] = ld8(vrow + s0 + 16 + half * 8);
            bv1[jt].h[0] = ld8(vrow + s0 + 32 + half * 8);
            bv1[jt].h[1] = ld8(vrow + s0 + 48 + half * 8);
        }

        // ---- scale, causal mask (diagonal tiles only), online softmax
        const bool needmask = (s0 + 63 > t0);
        float p[4][8];
#pragma unroll
        for (int i = 0; i < 8; ++i) {
            const int row = t0 + i + half * 8;
            float v0 = sc[0][i] * 0.125f;   // DH^-0.5
            float v1 = sc[1][i] * 0.125f;
            float v2 = sc[2][i] * 0.125f;
            float v3 = sc[3][i] * 0.125f;
            if (needmask) {
                if (s0 + ln      > row) v0 = -3.0e38f;
                if (s0 + 16 + ln > row) v1 = -3.0e38f;
                if (s0 + 32 + ln > row) v2 = -3.0e38f;
                if (s0 + 48 + ln > row) v3 = -3.0e38f;
            }
            float rmax = fmaxf(fmaxf(v0, v1), fmaxf(v2, v3));
            rmax = fmaxf(rmax, __shfl_xor(rmax, 1));
            rmax = fmaxf(rmax, __shfl_xor(rmax, 2));
            rmax = fmaxf(rmax, __shfl_xor(rmax, 4));
            rmax = fmaxf(rmax, __shfl_xor(rmax, 8));
            float mnew  = fmaxf(m_r[i], rmax);
            float alpha = __expf(m_r[i] - mnew);
            float e0 = __expf(v0 - mnew);
            float e1 = __expf(v1 - mnew);
            float e2 = __expf(v2 - mnew);
            float e3 = __expf(v3 - mnew);
            float rs = (e0 + e1) + (e2 + e3);
            rs += __shfl_xor(rs, 1);
            rs += __shfl_xor(rs, 2);
            rs += __shfl_xor(rs, 4);
            rs += __shfl_xor(rs, 8);
            l_r[i] = l_r[i] * alpha + rs;
            m_r[i] = mnew;
#pragma unroll
            for (int jt = 0; jt < 4; ++jt) o[jt][i] *= alpha;
            p[0][i] = e0; p[1][i] = e1; p[2][i] = e2; p[3][i] = e3;
        }

        // ---- C-layout -> A-layout through per-wave private LDS tile
#pragma unroll
        for (int i = 0; i < 8; ++i) {
            const int r = i + half * 8;
            plds[w][r][ln]      = (__bf16)p[0][i];
            plds[w][r][16 + ln] = (__bf16)p[1][i];
            plds[w][r][32 + ln] = (__bf16)p[2][i];
            plds[w][r][48 + ln] = (__bf16)p[3][i];
        }
        BFrag ap0, ap1;   // DS ops are in-order within a wave
        ap0.h[0] = *(const v8bf*)&plds[w][ln][half * 8];
        ap0.h[1] = *(const v8bf*)&plds[w][ln][16 + half * 8];
        ap1.h[0] = *(const v8bf*)&plds[w][ln][32 + half * 8];
        ap1.h[1] = *(const v8bf*)&plds[w][ln][48 + half * 8];

        // ---- O += P * V
#pragma unroll
        for (int jt = 0; jt < 4; ++jt) {
            o[jt] = WMMA_BF16(ap0.v, bv0[jt].v, o[jt]);
            o[jt] = WMMA_BF16(ap1.v, bv1[jt].v, o[jt]);
        }
    }

    // ---- normalize and store [B,T,H*DH] bf16
#pragma unroll
    for (int i = 0; i < 8; ++i) {
        float inv = 1.0f / l_r[i];
        const size_t trow = (size_t)(b * T_ + t0 + i + half * 8) * D_ + h * DH_;
#pragma unroll
        for (int jt = 0; jt < 4; ++jt)
            attn[trow + jt * 16 + ln] = (__bf16)(o[jt][i] * inv);
    }
}

// ---------------------------------------------------------------------------
// Host-side orchestration
// ---------------------------------------------------------------------------
extern "C" void kernel_launch(void* const* d_in, const int* in_sizes, int n_in,
                              void* d_out, int out_size, void* d_ws, size_t ws_size,
                              hipStream_t stream) {
    const float* x     = (const float*)d_in[0];   // [B,T,D]
    const float* w_qkv = (const float*)d_in[1];   // [D,3D]
    const float* w_out = (const float*)d_in[2];   // [D,D]
    float* out = (float*)d_out;                   // [B,T,D] f32

    char* ws = (char*)d_ws;
    size_t off = 0;
    __bf16* xb    = (__bf16*)(ws + off); off += (size_t)MROWS * D_   * 2;  // 8 MB
    __bf16* wqkvT = (__bf16*)(ws + off); off += (size_t)TD3  * D_    * 2;  // 6 MB
    __bf16* woutT = (__bf16*)(ws + off); off += (size_t)D_   * D_    * 2;  // 2 MB
    __bf16* qkvb  = (__bf16*)(ws + off); off += (size_t)MROWS * TD3  * 2;  // 24 MB
    __bf16* vtb   = (__bf16*)(ws + off); off += (size_t)B_ * H_ * DH_ * T_ * 2; // 8 MB
    __bf16* attnb = (__bf16*)(ws + off); off += (size_t)MROWS * D_   * 2;  // 8 MB
    (void)off; (void)ws_size; (void)in_sizes; (void)n_in; (void)out_size;

    // 1) convert inputs to bf16 (weights transposed to [N,K])
    cvt_f32_bf16<<<(MROWS * D_) / 256, 256, 0, stream>>>(x, xb, MROWS * D_);
    transpose_cvt<<<(D_ * TD3) / 256, 256, 0, stream>>>(w_qkv, wqkvT, D_, TD3);
    transpose_cvt<<<(D_ * D_) / 256, 256, 0, stream>>>(w_out, woutT, D_, D_);

    // 2) QKV projection: [4096,1024] x [1024,3072]
    gemm_bf16_wmma<<<dim3(TD3 / 128, MROWS / 64), 256, 0, stream>>>(
        xb, wqkvT, qkvb, nullptr, D_, TD3);

    // 3) RoPE on q,k
    rope_kernel<<<(B_ * T_ * H_ * 32) / 256, 256, 0, stream>>>(qkvb);

    // 4) transpose V for the PV matmul
    vtrans_kernel<<<(B_ * H_ * DH_ * T_) / 256, 256, 0, stream>>>(qkvb, vtb);

    // 5) fused causal flash attention
    attn_kernel<<<dim3(T_ / 64, B_ * H_), 128, 0, stream>>>(qkvb, vtb, attnb);

    // 6) output projection: [4096,1024] x [1024,1024] -> f32 out
    gemm_bf16_wmma<<<dim3(D_ / 128, MROWS / 64), 256, 0, stream>>>(
        attnb, woutT, nullptr, out, D_, D_);
}
